// DynamicCombiner_71141838291071
// MI455X (gfx1250) — matrix operand order repaired and checked by reading, not compiled, not run on hardware
//
#include <hip/hip_runtime.h>

// Problem constants (B=2, S=1024, D=1024, V=32000, K=16)
#define NROWS 2048
#define DDIM  1024
#define VDIM  32000
#define KNN   16

typedef __attribute__((ext_vector_type(16))) __bf16 v16bf;
typedef __attribute__((ext_vector_type(8)))  __bf16 v8bf;
typedef __attribute__((ext_vector_type(8)))  float  v8f;

__device__ __forceinline__ unsigned short f2bf(float x) {
  union { float f; unsigned u; } c; c.f = x;
  unsigned u = c.u;
  return (unsigned short)((u + 0x7FFFu + ((u >> 16) & 1u)) >> 16);  // RNE
}

__device__ __forceinline__ float block_sum(float v, float* red) {
  int t = threadIdx.x;
  red[t] = v; __syncthreads();
  for (int s = 128; s > 0; s >>= 1) {
    if (t < s) red[t] += red[t + s];
    __syncthreads();
  }
  float r = red[0]; __syncthreads();
  return r;
}

__device__ __forceinline__ float block_max(float v, float* red) {
  int t = threadIdx.x;
  red[t] = v; __syncthreads();
  for (int s = 128; s > 0; s >>= 1) {
    if (t < s) red[t] = fmaxf(red[t], red[t + s]);
    __syncthreads();
  }
  float r = red[0]; __syncthreads();
  return r;
}

// Kernel 1: per-row bandwidth, sparse_probs, merged; emit A = [h; merged] in bf16.
__global__ __launch_bounds__(256)
void row_prep(const float* __restrict__ hidden,
              const float* __restrict__ distances,
              const float* __restrict__ sh,
              const float* __restrict__ bW,
              const float* __restrict__ bb,
              unsigned short* __restrict__ Abf,
              float* __restrict__ sp_out) {
  const int n = blockIdx.x;
  const int t = threadIdx.x;
  __shared__ float red[256];
  __shared__ float sp_l[KNN];

  const float4* h4  = (const float4*)(hidden + (size_t)n * DDIM);
  const float4* w1  = (const float4*)(bW);
  const float4* w2  = (const float4*)(bW + DDIM);
  const float4* s4  = (const float4*)(sh + (size_t)n * KNN * DDIM);

  // h part: 256 quads, one per thread; also write bf16 h into A.
  float dot;
  {
    float4 hv = h4[t];
    float4 wv = w1[t];
    dot = hv.x * wv.x + hv.y * wv.y + hv.z * wv.z + hv.w * wv.w;
    unsigned short* ap = Abf + (size_t)n * (2 * DDIM) + t * 4;
    ap[0] = f2bf(hv.x); ap[1] = f2bf(hv.y); ap[2] = f2bf(hv.z); ap[3] = f2bf(hv.w);
  }
  // mean_h part of the bandwidth dot: (1/K) * sum_k sum_d sh[k,d]*bW[D+d]
  float dot2 = 0.f;
  for (int i = t; i < KNN * (DDIM / 4); i += 256) {
    float4 sv = s4[i];
    float4 wv = w2[i & (DDIM / 4 - 1)];
    dot2 += sv.x * wv.x + sv.y * wv.y + sv.z * wv.z + sv.w * wv.w;
  }
  dot += dot2 * (1.0f / (float)KNN);
  float lin = block_sum(dot, red) + bb[0];

  if (t == 0) {
    float bw  = __expf(lin);
    float inv = 1.0f / bw;
    float x[KNN]; float m = -3.4e38f;
    for (int k = 0; k < KNN; ++k) {
      x[k] = -distances[n * KNN + k] * inv;
      m = fmaxf(m, x[k]);
    }
    float s = 0.f;
    for (int k = 0; k < KNN; ++k) { x[k] = __expf(x[k] - m); s += x[k]; }
    float is = 1.0f / s;
    for (int k = 0; k < KNN; ++k) {
      float p = x[k] * is;
      sp_l[k] = p;
      sp_out[n * KNN + k] = p;
    }
  }
  __syncthreads();

  // merged[d] = sum_k sp[k] * sh[n,k,d]  (second read of sh; L2-hot)
  float4 acc = make_float4(0.f, 0.f, 0.f, 0.f);
  for (int k = 0; k < KNN; ++k) {
    float p = sp_l[k];
    float4 sv = s4[k * (DDIM / 4) + t];
    acc.x += p * sv.x; acc.y += p * sv.y; acc.z += p * sv.z; acc.w += p * sv.w;
  }
  unsigned short* ap = Abf + (size_t)n * (2 * DDIM) + DDIM + t * 4;
  ap[0] = f2bf(acc.x); ap[1] = f2bf(acc.y); ap[2] = f2bf(acc.z); ap[3] = f2bf(acc.w);
}

// Kernel 2: fp32 -> bf16 weight conversion for mW1 (D x 2D).
__global__ __launch_bounds__(256)
void conv_w(const float* __restrict__ W, unsigned short* __restrict__ Wbf) {
  int i = (blockIdx.x * 256 + threadIdx.x) * 4;
  float4 v = *(const float4*)(W + i);
  unsigned short* p = Wbf + i;
  p[0] = f2bf(v.x); p[1] = f2bf(v.y); p[2] = f2bf(v.z); p[3] = f2bf(v.w);
}

// Kernel 3: H = relu(A(2048x2048,bf16) x mW1^T(2048x1024,bf16) + mb1), fp32 out.
// One 16x16 output tile per wave, 64 k-steps of v_wmma_f32_16x16x32_bf16.
__global__ __launch_bounds__(256)
void gemm_wmma(const unsigned short* __restrict__ Abf,
               const unsigned short* __restrict__ Wbf,
               const float* __restrict__ mb1,
               float* __restrict__ H) {
  const int lane = threadIdx.x & 31;
  const int wave = threadIdx.x >> 5;
  const int tile = blockIdx.x * 8 + wave;
  const int tilesN = DDIM / 16;           // 64
  const int tm = tile / tilesN;           // 0..127
  const int tn = tile - tm * tilesN;      // 0..63
  const int m0 = tm * 16, j0 = tn * 16;
  const int hi = lane >> 4;               // 0/1
  const int lo = lane & 15;

  const __bf16* A = (const __bf16*)Abf;
  const __bf16* W = (const __bf16*)Wbf;
  // A fragment lane base: row m0+lo, K chunks [base..base+7] and [base+8..base+15], base=16*hi
  const __bf16* arow = A + (size_t)(m0 + lo) * (2 * DDIM) + hi * 16;
  // B fragment lane base: column j0+lo (= mW1 row), K rows 16*hi + 0..15 contiguous
  const __bf16* brow = W + (size_t)(j0 + lo) * (2 * DDIM) + hi * 16;

  v8f acc = {};
#pragma unroll 4
  for (int k0 = 0; k0 < 2 * DDIM; k0 += 32) {
    v8bf a_lo = *(const v8bf*)(arow + k0);
    v8bf a_hi = *(const v8bf*)(arow + k0 + 8);
    v16bf a = __builtin_shufflevector(a_lo, a_hi,
                                      0, 1, 2, 3, 4, 5, 6, 7,
                                      8, 9, 10, 11, 12, 13, 14, 15);
    v16bf b = *(const v16bf*)(brow + k0);
    acc = __builtin_amdgcn_wmma_f32_16x16x32_bf16(
        false, a, false, b, (short)0, acc, false, false);
  }

  const float bias = mb1[j0 + lo];
#pragma unroll
  for (int r = 0; r < 8; ++r) {
    int m = m0 + r + hi * 8;
    float v = acc[r] + bias;
    H[(size_t)m * DDIM + j0 + lo] = v > 0.f ? v : 0.f;
  }
}

// Kernel 4: mixing[n] = sigmoid(H[n,:] . mW2 + mb2)
__global__ __launch_bounds__(256)
void mixing_k(const float* __restrict__ H, const float* __restrict__ mW2,
              const float* __restrict__ mb2, float* __restrict__ mixing) {
  const int n = blockIdx.x, t = threadIdx.x;
  __shared__ float red[256];
  const float4* h4 = (const float4*)(H + (size_t)n * DDIM);
  const float4* w4 = (const float4*)mW2;
  float4 hv = h4[t], wv = w4[t];
  float dot = hv.x * wv.x + hv.y * wv.y + hv.z * wv.z + hv.w * wv.w;
  float s = block_sum(dot, red) + mb2[0];
  if (t == 0) mixing[n] = 1.0f / (1.0f + __expf(-s));
}

// Kernel 5: log((1-mix)*softmax(lg) + mix*example_dist), sparse fix-up for the K indices.
__global__ __launch_bounds__(256)
void out_k(const float* __restrict__ logits, const long long* __restrict__ tok,
           const float* __restrict__ sp, const float* __restrict__ mixing,
           float* __restrict__ out) {
  const int n = blockIdx.x, t = threadIdx.x;
  __shared__ float red[256];
  const float* lg = logits + (size_t)n * VDIM;
  float* o = out + (size_t)n * VDIM;
  const float4* lg4 = (const float4*)lg;
  float4* o4 = (float4*)o;
  const int Q = VDIM / 4;  // 8000 quads

  // pass 1: row max
  float m = -3.4e38f;
  for (int i = t; i < Q; i += 256) {
    float4 v = lg4[i];
    m = fmaxf(m, fmaxf(fmaxf(v.x, v.y), fmaxf(v.z, v.w)));
  }
  m = block_max(m, red);

  // pass 2: sum exp
  float s = 0.f;
  for (int i = t; i < Q; i += 256) {
    float4 v = lg4[i];
    s += __expf(v.x - m) + __expf(v.y - m) + __expf(v.z - m) + __expf(v.w - m);
  }
  s = block_sum(s, red);

  const float mix = mixing[n];
  const float base = __logf(1.0f - mix) - m - __logf(s);

  // pass 3: dense part (example_dist == 0 almost everywhere)
  for (int i = t; i < Q; i += 256) {
    float4 v = lg4[i];
    o4[i] = make_float4(v.x + base, v.y + base, v.z + base, v.w + base);
  }
  __threadfence_block();
  __syncthreads();

  // sparse fix-up (duplicate indices sum their probs, matching .at[].add)
  if (t < KNN) {
    int idx = (int)tok[n * KNN + t];
    float e = 0.f;
    for (int k = 0; k < KNN; ++k)
      if ((int)tok[n * KNN + k] == idx) e += sp[n * KNN + k];
    float p = (1.0f - mix) * __expf(lg[idx] - m) / s + mix * e;
    o[idx] = __logf(p);
  }
}

extern "C" void kernel_launch(void* const* d_in, const int* in_sizes, int n_in,
                              void* d_out, int out_size, void* d_ws, size_t ws_size,
                              hipStream_t stream) {
  const float*     hidden    = (const float*)d_in[0];
  const float*     logits    = (const float*)d_in[1];
  const float*     distances = (const float*)d_in[2];
  const float*     sh        = (const float*)d_in[3];
  const long long* tok       = (const long long*)d_in[4];
  const float*     bW        = (const float*)d_in[5];
  const float*     bb        = (const float*)d_in[6];
  const float*     mW1       = (const float*)d_in[7];
  const float*     mb1       = (const float*)d_in[8];
  const float*     mW2       = (const float*)d_in[9];
  const float*     mb2       = (const float*)d_in[10];
  float* out = (float*)d_out;

  char* ws = (char*)d_ws;
  unsigned short* Abf = (unsigned short*)(ws);                      // 2048*2048*2 = 8 MB
  unsigned short* Wbf = (unsigned short*)(ws + (size_t)(8u << 20)); // 1024*2048*2 = 4 MB
  float* sp     = (float*)(ws + (size_t)(12u << 20));               // 2048*16*4   = 128 KB
  float* H      = (float*)(ws + (size_t)(13u << 20));               // 2048*1024*4 = 8 MB
  float* mixing = (float*)(ws + (size_t)(21u << 20));               // 2048*4      = 8 KB

  row_prep<<<NROWS, 256, 0, stream>>>(hidden, distances, sh, bW, bb, Abf, sp);
  conv_w<<<(DDIM * 2 * DDIM) / (4 * 256), 256, 0, stream>>>(mW1, Wbf);
  gemm_wmma<<<(NROWS / 16) * (DDIM / 16) / 8, 256, 0, stream>>>(Abf, Wbf, mb1, H);
  mixing_k<<<NROWS, 256, 0, stream>>>(H, mW2, mb2, mixing);
  out_k<<<NROWS, 256, 0, stream>>>(logits, tok, sp, mixing, out);
}
